// SelfAttention_19086834663492
// MI455X (gfx1250) — matrix operand order repaired and checked by reading, not compiled, not run on hardware
//
#include <hip/hip_runtime.h>

// ---------------------------------------------------------------------------
// Self-attention (B=8, N=4096, C=256, D=32) for gfx1250 (MI455X, wave32, WMMA)
//   kernel 1: fused QKV projection  x[32768,256] x W[256,32] (x3) -> bf16 ws
//             (V is stored TRANSPOSED per batch: [D][N], token-contiguous)
//   kernel 2: flash attention (BM=128 rows/WG, BN=64 keys/tile) with K / V^T
//             tiles fetched by the Tensor Data Mover (tensor_load_to_lds,
//             TENSORcnt), then O@Wo + bo, gamma*o + x residual.
//             All matmuls: v_wmma_f32_16x16x32_bf16.
// ---------------------------------------------------------------------------

typedef __bf16 bf16;
typedef __attribute__((ext_vector_type(16))) __bf16 v16bf;
typedef __attribute__((ext_vector_type(8)))  __bf16 v8bf;
typedef __attribute__((ext_vector_type(8)))  float  v8f;
typedef __attribute__((ext_vector_type(8)))  float  f32x8;
typedef __attribute__((ext_vector_type(4)))  unsigned int v4u;
typedef __attribute__((ext_vector_type(8)))  int    v8i;
typedef __attribute__((ext_vector_type(4)))  int    v4i;

constexpr int BATCH = 8;
constexpr int NTOK  = 4096;    // 64*64
constexpr int CIN   = 256;
constexpr int DH    = 32;      // head dim == WMMA K depth for bf16
constexpr int BM    = 128;     // query rows per workgroup (8 waves x 16)
constexpr int BN    = 64;      // keys per tile

__device__ __forceinline__ v8f v8f_zero() {
  v8f z;
#pragma unroll
  for (int i = 0; i < 8; ++i) z[i] = 0.0f;
  return z;
}

// reduce across the 16-lane half of a wave32 (xor masks 1,2,4,8)
__device__ __forceinline__ float half16_max(float v) {
  v = fmaxf(v, __builtin_bit_cast(float, __builtin_amdgcn_ds_swizzle(__builtin_bit_cast(int, v), 0x041F)));
  v = fmaxf(v, __builtin_bit_cast(float, __builtin_amdgcn_ds_swizzle(__builtin_bit_cast(int, v), 0x081F)));
  v = fmaxf(v, __builtin_bit_cast(float, __builtin_amdgcn_ds_swizzle(__builtin_bit_cast(int, v), 0x101F)));
  v = fmaxf(v, __builtin_bit_cast(float, __builtin_amdgcn_ds_swizzle(__builtin_bit_cast(int, v), 0x201F)));
  return v;
}
__device__ __forceinline__ float half16_sum(float v) {
  v += __builtin_bit_cast(float, __builtin_amdgcn_ds_swizzle(__builtin_bit_cast(int, v), 0x041F));
  v += __builtin_bit_cast(float, __builtin_amdgcn_ds_swizzle(__builtin_bit_cast(int, v), 0x081F));
  v += __builtin_bit_cast(float, __builtin_amdgcn_ds_swizzle(__builtin_bit_cast(int, v), 0x101F));
  v += __builtin_bit_cast(float, __builtin_amdgcn_ds_swizzle(__builtin_bit_cast(int, v), 0x201F));
  return v;
}

// ---------------------------------------------------------------------------
// TDM: issue a 2D tensor_load_to_lds per CDNA5 ISA
//   element size = 2 bytes (bf16); tile [tile_d1 rows][tile_d0 cols] from a
//   tensor whose innermost dim is tensor_d0 with row stride stride0 (elems).
// ---------------------------------------------------------------------------
__device__ __forceinline__ void tdm_load_2d_bf16(unsigned lds_addr,
                                                 const void* gaddr,
                                                 unsigned tensor_d0, unsigned tensor_d1,
                                                 unsigned tile_d0,   unsigned tile_d1,
                                                 unsigned long long stride0) {
  unsigned long long ga = (unsigned long long)(size_t)gaddr;
  v4u g0;
  g0[0] = 1u;                                       // count=1, user descriptor
  g0[1] = lds_addr;                                 // lds_addr [63:32]
  g0[2] = (unsigned)(ga & 0xffffffffu);             // global_addr [95:64]
  g0[3] = (unsigned)((ga >> 32) & 0x01ffffffu)      // global_addr [120:96]
        | 0x80000000u;                              // type=2 ("image") [127:126]
  v8i g1;
  g1[0] = (int)(1u << 16);                          // data_size=1 (2 bytes)
  g1[1] = (int)((tensor_d0 & 0xffffu) << 16);       // tensor_dim0 [79:48]
  g1[2] = (int)(((tensor_d0 >> 16) & 0xffffu) | ((tensor_d1 & 0xffffu) << 16));
  g1[3] = (int)(((tensor_d1 >> 16) & 0xffffu) | ((tile_d0 & 0xffffu) << 16));
  g1[4] = (int)(tile_d1 & 0xffffu);                 // tile_dim1; tile_dim2=0
  g1[5] = (int)(stride0 & 0xffffffffu);             // tensor_dim0_stride [207:160]
  g1[6] = (int)((stride0 >> 32) & 0xffffu);         // dim1_stride=0
  g1[7] = 0;
  v4i gz = {0, 0, 0, 0};                            // groups 2/3 unused (2D)
#if __has_include(<hip/amd_detail/amd_gfx1250_TDM.h>)
  v8i gz8 = {0, 0, 0, 0, 0, 0, 0, 0};
  __builtin_amdgcn_tensor_load_to_lds(g0, g1, gz, gz, gz8, 0);   // clang-23 form
#else
  __builtin_amdgcn_tensor_load_to_lds(g0, g1, gz, gz, 0);        // ROCm 7.2 form
#endif
}

// ---------------------------------------------------------------------------
// Kernel 1: QKV projection. 128 threads (4 waves), 64 rows of x per block.
// Weights staged transposed in LDS as bf16: WT[w][n_out][k] (48 KB).
// F(keys) and G(queries) stored [B*N][D]; V stored transposed [B][D][N].
// ---------------------------------------------------------------------------
__global__ void __launch_bounds__(128)
qkv_kernel(const float* __restrict__ x,
           const float* __restrict__ Wf, const float* __restrict__ bfv,
           const float* __restrict__ Wg, const float* __restrict__ bgv,
           const float* __restrict__ Wh, const float* __restrict__ bhv,
           bf16* __restrict__ Fk, bf16* __restrict__ Gq, bf16* __restrict__ VvT)
{
  __shared__ __align__(64) bf16 WT[3][DH][CIN];   // 48 KB

  const float* Ws[3] = {Wf, Wg, Wh};
#pragma unroll
  for (int w = 0; w < 3; ++w)
    for (int idx = threadIdx.x; idx < CIN * DH; idx += 128) {
      int k = idx >> 5, n = idx & 31;             // W is [C,D] row-major
      WT[w][n][k] = (bf16)Ws[w][idx];
    }
  __syncthreads();

  const int lane = threadIdx.x & 31;
  const int wave = threadIdx.x >> 5;
  const int n16  = lane & 15;
  const int hi   = lane >> 4;
  const int rowBase = blockIdx.x * 64 + wave * 16;

  v8f acc[3][2];
#pragma unroll
  for (int w = 0; w < 3; ++w) { acc[w][0] = v8f_zero(); acc[w][1] = v8f_zero(); }

  const float* xr = x + (size_t)(rowBase + n16) * CIN;

#pragma unroll
  for (int ks = 0; ks < CIN / 32; ++ks) {
    // A fragment (16x32 bf16): lane m=n16, K chunks [hi*16,+8) and [hi*16+8,+8)
    f32x8 c0 = *(const f32x8*)(xr + ks * 32 + hi * 16);
    f32x8 c1 = *(const f32x8*)(xr + ks * 32 + hi * 16 + 8);
    v16bf a;
#pragma unroll
    for (int i = 0; i < 8; ++i) { a[i] = (bf16)c0[i]; a[8 + i] = (bf16)c1[i]; }

#pragma unroll
    for (int w = 0; w < 3; ++w)
#pragma unroll
      for (int t = 0; t < 2; ++t) {
        v16bf bm = *(const v16bf*)&WT[w][t * 16 + n16][ks * 32 + hi * 16];
        acc[w][t] = __builtin_amdgcn_wmma_f32_16x16x32_bf16(
            false, a, false, bm, (short)0, acc[w][t], false, false);
      }
  }

  // F, G: row-major [token][d]
  const float* biases2[2] = {bfv, bgv};
  bf16* outs2[2] = {Fk, Gq};
#pragma unroll
  for (int w = 0; w < 2; ++w)
#pragma unroll
    for (int t = 0; t < 2; ++t) {
      int col = t * 16 + n16;
      float bias = biases2[w][col];
#pragma unroll
      for (int r = 0; r < 8; ++r) {
        int row = rowBase + r + 8 * hi;            // C/D layout
        outs2[w][(size_t)row * DH + col] = (bf16)(acc[w][t][r] + bias);
      }
    }
  // V: transposed per batch [d][token] so TDM can tile it directly
#pragma unroll
  for (int t = 0; t < 2; ++t) {
    int col = t * 16 + n16;
    float bias = bhv[col];
#pragma unroll
    for (int r = 0; r < 8; ++r) {
      int rowg = rowBase + r + 8 * hi;
      int bb = rowg >> 12, nn = rowg & (NTOK - 1);
      VvT[((size_t)bb * DH + col) * NTOK + nn] = (bf16)(acc[2][t][r] + bias);
    }
  }
}

// ---------------------------------------------------------------------------
// Kernel 2: flash attention + output projection + residual.
// 256 threads (8 waves); each wave owns 16 query rows; loop over 64-key tiles
// staged into LDS by the Tensor Data Mover.
// ---------------------------------------------------------------------------
__global__ void __launch_bounds__(256)
attn_kernel(const bf16* __restrict__ Fk, const bf16* __restrict__ Gq,
            const bf16* __restrict__ VvT,
            const float* __restrict__ Wo, const float* __restrict__ bo,
            const float* __restrict__ gamma_p,
            const float* __restrict__ x, float* __restrict__ out)
{
  __shared__ __align__(64) bf16 Kt[BN * DH];        //  4 KB  keys   [key][d]
  __shared__ __align__(64) bf16 Vt[DH * BN];        //  4 KB  values [d][key]
  __shared__ __align__(64) bf16 WoT[CIN * DH];      // 16 KB  Wo transposed [c][d]
  __shared__ __align__(64) bf16 Pl[8][16 * BN];     // 16 KB  per-wave P / O bounce

  const int tid  = threadIdx.x;
  const int lane = tid & 31;
  const int wave = tid >> 5;
  const int n16  = lane & 15;
  const int hi   = lane >> 4;
  const int b    = blockIdx.y;
  const int rowBase = blockIdx.x * BM + wave * 16;
  const size_t batchOff = (size_t)b * NTOK * DH;

  // stage Wo (32x256 f32 row-major) transposed as bf16
  for (int idx = tid; idx < CIN * DH; idx += 256) {
    int d = idx >> 8, c = idx & 255;
    WoT[c * DH + d] = (bf16)Wo[idx];
  }

  // Q fragment for this wave's 16 rows (A layout), stays in registers
  const bf16* qr = Gq + batchOff + (size_t)(rowBase + n16) * DH;
  v8bf q0 = *(const v8bf*)(qr + hi * 16);
  v8bf q1 = *(const v8bf*)(qr + hi * 16 + 8);
  v16bf qfrag = __builtin_shufflevector(q0, q1, 0, 1, 2, 3, 4, 5, 6, 7,
                                        8, 9, 10, 11, 12, 13, 14, 15);

  v8f o0 = v8f_zero(), o1 = v8f_zero();
  v8f m_run, l_run;
#pragma unroll
  for (int r = 0; r < 8; ++r) { m_run[r] = -3.0e38f; l_run[r] = 0.0f; }

  bf16* pw = &Pl[wave][0];
  const unsigned ldsK = (unsigned)(size_t)&Kt[0];
  const unsigned ldsV = (unsigned)(size_t)&Vt[0];

  for (int kb = 0; kb < NTOK / BN; ++kb) {
    const size_t base = batchOff + (size_t)kb * BN * DH;

    // --- TDM: one wave issues both 2D tile DMAs; others wait at barrier ---
    if (tid < 32) {
      // K tile: tensor [NTOK][DH] (d contiguous), tile [BN][DH] -> Kt[64][32]
      tdm_load_2d_bf16(ldsK, Fk + base, DH, NTOK, DH, BN, DH);
      // V^T tile: tensor [DH][NTOK] (token contiguous), tile [DH][BN] -> Vt[32][64]
      tdm_load_2d_bf16(ldsV, VvT + (size_t)b * DH * NTOK + (size_t)kb * BN,
                       NTOK, DH, BN, DH, NTOK);
      __builtin_amdgcn_s_wait_tensorcnt(0);
    }
    __syncthreads();

    if (kb + 1 < NTOK / BN)
      __builtin_prefetch(Fk + base + (size_t)BN * DH, 0, 1);  // global_prefetch_b8

    // S tile: 16 x 64 logits = 4 WMMAs (K depth = DH = 32 exactly)
    v8f s[4];
#pragma unroll
    for (int t = 0; t < 4; ++t) {
      v16bf kf = *(const v16bf*)&Kt[(t * 16 + n16) * DH + hi * 16];
      s[t] = __builtin_amdgcn_wmma_f32_16x16x32_bf16(
          false, qfrag, false, kf, (short)0, v8f_zero(), false, false);
    }

    // ---- online softmax (fp32) ----
    v8f mnew;
#pragma unroll
    for (int r = 0; r < 8; ++r) {
      float mx = fmaxf(fmaxf(s[0][r], s[1][r]), fmaxf(s[2][r], s[3][r]));
      mx = half16_max(mx);                   // over the row's 16 lanes
      mnew[r] = fmaxf(m_run[r], mx);
    }
    v8f scale;
#pragma unroll
    for (int r = 0; r < 8; ++r) scale[r] = __expf(m_run[r] - mnew[r]);
    o0 *= scale; o1 *= scale;

    v8f rsum = v8f_zero();
#pragma unroll
    for (int t = 0; t < 4; ++t)
#pragma unroll
      for (int r = 0; r < 8; ++r) {
        float p = __expf(s[t][r] - mnew[r]);
        s[t][r] = p;
        rsum[r] += p;
      }
#pragma unroll
    for (int r = 0; r < 8; ++r) rsum[r] = half16_sum(rsum[r]);
    l_run = l_run * scale + rsum;
    m_run = mnew;

    // P (C layout) -> per-wave LDS row-major [16][64] as bf16
#pragma unroll
    for (int t = 0; t < 4; ++t)
#pragma unroll
      for (int r = 0; r < 8; ++r)
        pw[(r + 8 * hi) * BN + t * 16 + n16] = (bf16)s[t][r];

    // O += P @ V : 2 K-steps x 2 d-tiles = 4 WMMAs
#pragma unroll
    for (int ks = 0; ks < 2; ++ks) {
      v8bf p0 = *(const v8bf*)&pw[n16 * BN + ks * 32 + hi * 16];
      v8bf p1 = *(const v8bf*)&pw[n16 * BN + ks * 32 + hi * 16 + 8];
      v16bf pf = __builtin_shufflevector(p0, p1, 0, 1, 2, 3, 4, 5, 6, 7,
                                         8, 9, 10, 11, 12, 13, 14, 15);
      v16bf vf0 = *(const v16bf*)&Vt[(0 * 16 + n16) * BN + ks * 32 + hi * 16];
      v16bf vf1 = *(const v16bf*)&Vt[(1 * 16 + n16) * BN + ks * 32 + hi * 16];
      o0 = __builtin_amdgcn_wmma_f32_16x16x32_bf16(false, pf, false, vf0,
                                                   (short)0, o0, false, false);
      o1 = __builtin_amdgcn_wmma_f32_16x16x32_bf16(false, pf, false, vf1,
                                                   (short)0, o1, false, false);
    }
    __syncthreads();
  }

  // normalize
  v8f inv;
#pragma unroll
  for (int r = 0; r < 8; ++r) inv[r] = 1.0f / l_run[r];
  o0 *= inv; o1 *= inv;

  // O (C layout 16x32) -> LDS -> A layout fragment
#pragma unroll
  for (int r = 0; r < 8; ++r) {
    pw[(r + 8 * hi) * DH + n16]      = (bf16)o0[r];
    pw[(r + 8 * hi) * DH + 16 + n16] = (bf16)o1[r];
  }
  v8bf a0 = *(const v8bf*)&pw[n16 * DH + hi * 16];
  v8bf a1 = *(const v8bf*)&pw[n16 * DH + hi * 16 + 8];
  v16bf ofrag = __builtin_shufflevector(a0, a1, 0, 1, 2, 3, 4, 5, 6, 7,
                                        8, 9, 10, 11, 12, 13, 14, 15);

  const float gamma = gamma_p[0];
  // out tile 16 x 256 = 16 WMMAs (K depth = DH = 32)
#pragma unroll
  for (int t = 0; t < CIN / 16; ++t) {
    v16bf wf = *(const v16bf*)&WoT[(t * 16 + n16) * DH + hi * 16];
    v8f acc = __builtin_amdgcn_wmma_f32_16x16x32_bf16(
        false, ofrag, false, wf, (short)0, v8f_zero(), false, false);
    int col = t * 16 + n16;
    float bias = bo[col];
#pragma unroll
    for (int r = 0; r < 8; ++r) {
      int row = rowBase + r + 8 * hi;
      size_t gi = ((size_t)b * NTOK + row) * CIN + col;
      out[gi] = gamma * (acc[r] + bias) + x[gi];
    }
  }
}

// ---------------------------------------------------------------------------
extern "C" void kernel_launch(void* const* d_in, const int* in_sizes, int n_in,
                              void* d_out, int out_size, void* d_ws, size_t ws_size,
                              hipStream_t stream) {
  (void)in_sizes; (void)n_in; (void)out_size; (void)ws_size;
  const float* x   = (const float*)d_in[0];
  const float* Wf  = (const float*)d_in[1];
  const float* bfv = (const float*)d_in[2];
  const float* Wg  = (const float*)d_in[3];
  const float* bgv = (const float*)d_in[4];
  const float* Wh  = (const float*)d_in[5];
  const float* bhv = (const float*)d_in[6];
  const float* Wo  = (const float*)d_in[7];
  const float* bo  = (const float*)d_in[8];
  const float* gam = (const float*)d_in[9];

  const size_t per = (size_t)BATCH * NTOK * DH;   // 1,048,576 elems
  bf16* Fk  = (bf16*)d_ws;                        // keys    (x @ Wf), [B*N][D]
  bf16* Gq  = Fk + per;                           // queries (x @ Wg), [B*N][D]
  bf16* VvT = Gq + per;                           // values  (x @ Wh), [B][D][N]

  qkv_kernel<<<(BATCH * NTOK) / 64, 128, 0, stream>>>(
      x, Wf, bfv, Wg, bgv, Wh, bhv, Fk, Gq, VvT);

  attn_kernel<<<dim3(NTOK / BM, BATCH), 256, 0, stream>>>(
      Fk, Gq, VvT, Wo, bo, gam, x, (float*)d_out);
}
